// Locoformer_36670430773896
// MI455X (gfx1250) — compile-verified, matrix operand order checked by reading, not run on hardware
//
#include <hip/hip_runtime.h>
#include <hip/hip_bf16.h>
#include <math.h>

// ---------------- problem constants (match reference) ----------------
#define B_    2
#define S_    2048
#define DIM_  1024
#define H_    16
#define DH_   64
#define L_    4
#define WIN_  512
#define DIN_  2730           // int(1024*4.0*2/3)
#define KP2_  2752           // DIN_ rounded up to 32 (K stride for w2/g2)
#define HD_   (H_ * DH_)     // 1024
#define M_    (B_ * S_)      // 4096 rows
#define EPS_  1.1920929e-07f
#define SCALE_ 0.125f        // 64^-0.5

typedef __attribute__((ext_vector_type(16))) __bf16 v16bf;
typedef __attribute__((ext_vector_type(8)))  float  v8f;

// fp32 -> bf16 round-to-nearest-even
__device__ __forceinline__ unsigned short f2bf(float f) {
  unsigned int u = __builtin_bit_cast(unsigned int, f);
  u = (u + 0x7FFFu + ((u >> 16) & 1u)) >> 16;
  return (unsigned short)u;
}
// ISA 7.12.2: 16-bit A/B fragment K mapping. VGPR i, lane-half -> K pair base.
__device__ __forceinline__ int kmap(int i, int half) {
  return ((i & 4) ? 16 : 0) + half * 8 + (i & 3) * 2;
}

// =====================================================================
// one-time weight transpose + fp32->bf16: in w[K][N] -> out[Npad][Kpad]
// rows n>=N and cols k>=K are written as zero.
// =====================================================================
__global__ __launch_bounds__(256)
void convert_wT_k(const float* __restrict__ w, unsigned short* __restrict__ out,
                  int K, int N, int Kpad)
{
  __shared__ unsigned short T[32][33];
  const int kt = blockIdx.x * 32;
  const int nt = blockIdx.y * 32;
  const int tx = threadIdx.x & 31;
  const int ty = threadIdx.x >> 5;   // 0..7
  #pragma unroll
  for (int i = 0; i < 4; ++i) {
    int k = kt + ty + i * 8;
    int n = nt + tx;
    float v = (k < K && n < N) ? w[(size_t)k * N + n] : 0.0f;
    T[ty + i * 8][tx] = f2bf(v);     // T[k_local][n_local]
  }
  __syncthreads();
  #pragma unroll
  for (int i = 0; i < 4; ++i) {
    int n = nt + ty + i * 8;
    int k = kt + tx;
    out[(size_t)n * Kpad + k] = T[tx][ty + i * 8];
  }
}

// =====================================================================
// LDS-free bf16 WMMA GEMM. A[M][lda] bf16 (K-major), Bt[Npad][ldb] bf16
// (K-major, transposed weights). C fp32. EPI bits: 1=+bias 2=+res 4=sigmoid.
// Block = 256 thr = 8 waves, tile 64(M) x 128(N); each wave 32x32 (2x2 WMMA).
// Fragment = two global_load_b128 per 16-row operand (ISA 7.12.2 layout).
// =====================================================================
#define GBM 64
#define GBN 128

template <int EPI>
__global__ __launch_bounds__(256)
void gemm_bf16(const unsigned short* __restrict__ A,
               const unsigned short* __restrict__ Bt,
               const float* __restrict__ bias, const float* __restrict__ res,
               float* __restrict__ C, int M, int N, int K, int lda, int ldb)
{
  const int lane  = threadIdx.x & 31;
  const int half  = lane >> 4;
  const int l15   = lane & 15;
  const int wv    = threadIdx.x >> 5;
  const int waveM = wv >> 2;    // 0..1
  const int waveN = wv & 3;     // 0..3
  const int bm = blockIdx.y * GBM;
  const int bn = blockIdx.x * GBN;

  const unsigned short* pa0 = A  + (size_t)(bm + waveM * 32 + l15) * lda + half * 8;
  const unsigned short* pa1 = pa0 + (size_t)16 * lda;
  const unsigned short* pb0 = Bt + (size_t)(bn + waveN * 32 + l15) * ldb + half * 8;
  const unsigned short* pb1 = pb0 + (size_t)16 * ldb;

  union F { v16bf v; uint4 q[2]; };
  v8f acc[2][2] = {};

  for (int k0 = 0; k0 < K; k0 += 32) {
    __builtin_prefetch(pa0 + k0 + 512, 0, 0);
    __builtin_prefetch(pa1 + k0 + 512, 0, 0);
    __builtin_prefetch(pb0 + k0 + 512, 0, 0);
    __builtin_prefetch(pb1 + k0 + 512, 0, 0);
    F a0, a1, b0, b1;
    a0.q[0] = *(const uint4*)(pa0 + k0);
    a0.q[1] = *(const uint4*)(pa0 + k0 + 16);
    a1.q[0] = *(const uint4*)(pa1 + k0);
    a1.q[1] = *(const uint4*)(pa1 + k0 + 16);
    b0.q[0] = *(const uint4*)(pb0 + k0);
    b0.q[1] = *(const uint4*)(pb0 + k0 + 16);
    b1.q[0] = *(const uint4*)(pb1 + k0);
    b1.q[1] = *(const uint4*)(pb1 + k0 + 16);
    acc[0][0] = __builtin_amdgcn_wmma_f32_16x16x32_bf16(false, a0.v, false, b0.v,
                                                        (short)0, acc[0][0], false, false);
    acc[0][1] = __builtin_amdgcn_wmma_f32_16x16x32_bf16(false, a0.v, false, b1.v,
                                                        (short)0, acc[0][1], false, false);
    acc[1][0] = __builtin_amdgcn_wmma_f32_16x16x32_bf16(false, a1.v, false, b0.v,
                                                        (short)0, acc[1][0], false, false);
    acc[1][1] = __builtin_amdgcn_wmma_f32_16x16x32_bf16(false, a1.v, false, b1.v,
                                                        (short)0, acc[1][1], false, false);
  }

  #pragma unroll
  for (int msub = 0; msub < 2; ++msub) {
    #pragma unroll
    for (int ct = 0; ct < 2; ++ct) {
      int col = bn + waveN * 32 + ct * 16 + l15;
      if (col < N) {
        #pragma unroll
        for (int r = 0; r < 8; ++r) {
          int row = bm + waveM * 32 + msub * 16 + r + half * 8;   // C layout
          float vv = acc[msub][ct][r];
          if (EPI & 1) vv += bias[col];
          if (EPI & 4) vv = 1.0f / (1.0f + __expf(-vv));
          if (EPI & 2) vv += res[(size_t)row * N + col];
          C[(size_t)row * N + col] = vv;
        }
      }
    }
  }
}

// =====================================================================
// flash sliding-window attention, 1 wave per 16-query tile per (b,h).
// q,k bf16 [B,S,H,DH]; v bf16 TRANSPOSED [B,H,DH,S]; out bf16 [B,S,H,DH].
// =====================================================================
__global__ __launch_bounds__(32)
void attn_wmma_k(const unsigned short* __restrict__ q,
                 const unsigned short* __restrict__ k,
                 const unsigned short* __restrict__ vt,
                 const float* __restrict__ gates,
                 unsigned short* __restrict__ out)
{
  __shared__ unsigned short Ps[16][34];       // P tile 16x32 bf16
  const int lane = threadIdx.x;
  const int half = lane >> 4;
  const int l15  = lane & 15;
  const int qb = blockIdx.x * 16;
  const int bh = blockIdx.y;
  const int b  = bh >> 4;                     // H_ == 16
  const int h  = bh & 15;

#define IDX(bb, ss, dd) ((((size_t)(bb) * S_ + (size_t)(ss)) * H_ + h) * DH_ + (dd))

  union FragU { v16bf v; unsigned int u[8]; };

  // Q A-fragments (q already RoPE'd and pre-scaled); bf16 pairs contiguous in d
  FragU faQ[2];
  {
    int m = qb + l15;
    #pragma unroll
    for (int c = 0; c < 2; ++c)
      #pragma unroll
      for (int i = 0; i < 8; ++i) {
        int d = c * 32 + kmap(i, half);
        faQ[c].u[i] = *(const unsigned int*)&q[IDX(b, m, d)];
      }
  }

  v8f oacc[4] = {};
  float rmax[8], rsum[8];
  #pragma unroll
  for (int r = 0; r < 8; ++r) { rmax[r] = -3.4028235e38f; rsum[r] = 0.0f; }

  const size_t vbase = ((size_t)b * H_ + h) * DH_ * S_;

  int kb0 = qb - WIN_; if (kb0 < 0) kb0 = 0; kb0 &= ~31;
  for (int kb = kb0; kb <= qb + 15; kb += 32) {
    // ---- sim = Q @ K^T, 32 keys as two 16-key n-tiles ----
    v8f sim[2] = {};
    #pragma unroll
    for (int nt = 0; nt < 2; ++nt) {
      int j  = kb + nt * 16 + l15;
      int jc = j >= S_ ? S_ - 1 : j;          // kb >= 0, clamp high side only
      #pragma unroll
      for (int c = 0; c < 2; ++c) {
        FragU fbK;
        #pragma unroll
        for (int i = 0; i < 8; ++i) {
          int d = c * 32 + kmap(i, half);
          fbK.u[i] = *(const unsigned int*)&k[IDX(b, jc, d)];
        }
        sim[nt] = __builtin_amdgcn_wmma_f32_16x16x32_bf16(false, faQ[c].v, false, fbK.v,
                                                          (short)0, sim[nt], false, false);
      }
    }
    // ---- mask + per-row block max (rows live in 16-lane half-groups) ----
    float bmax[8];
    #pragma unroll
    for (int r = 0; r < 8; ++r) {
      int m = qb + r + half * 8;
      #pragma unroll
      for (int nt = 0; nt < 2; ++nt) {
        int j = kb + nt * 16 + l15;
        int dist = m - j;
        bool valid = (dist >= 0) && (dist <= WIN_);
        sim[nt][r] = valid ? sim[nt][r] : -3.4028235e38f;
      }
      float mloc = fmaxf(sim[0][r], sim[1][r]);
      #pragma unroll
      for (int sh = 1; sh < 16; sh <<= 1)
        mloc = fmaxf(mloc, __shfl_xor(mloc, sh, 32));
      bmax[r] = mloc;
    }
    // ---- online softmax update ----
    #pragma unroll
    for (int r = 0; r < 8; ++r) {
      float nm = fmaxf(rmax[r], bmax[r]);
      float scale = __expf(rmax[r] - nm);
      rmax[r] = nm;
      float psum = 0.0f;
      #pragma unroll
      for (int nt = 0; nt < 2; ++nt) {
        float p = __expf(sim[nt][r] - nm);
        sim[nt][r] = p;
        psum += p;
      }
      #pragma unroll
      for (int sh = 1; sh < 16; sh <<= 1)
        psum += __shfl_xor(psum, sh, 32);
      rsum[r] = rsum[r] * scale + psum;
      #pragma unroll
      for (int t = 0; t < 4; ++t)
        oacc[t][r] *= scale;
    }
    // ---- C-layout P -> LDS -> A-layout fragment ----
    #pragma unroll
    for (int r = 0; r < 8; ++r) {
      int m = r + half * 8;
      Ps[m][l15]      = f2bf(sim[0][r]);
      Ps[m][16 + l15] = f2bf(sim[1][r]);
    }
    __syncthreads();
    FragU faP;
    #pragma unroll
    for (int i = 0; i < 8; ++i) {
      int kk = kmap(i, half);
      faP.u[i] = *(const unsigned int*)&Ps[l15][kk];
    }
    __syncthreads();
    // ---- O += P @ V; V transposed so key-pairs are contiguous dwords ----
    #pragma unroll
    for (int nt = 0; nt < 4; ++nt) {
      FragU fbV;
      int d = nt * 16 + l15;
      const unsigned short* vrow = vt + vbase + (size_t)d * S_;
      #pragma unroll
      for (int i = 0; i < 8; ++i) {
        int sc = kb + kmap(i, half);
        if (sc > S_ - 2) sc = S_ - 2;         // masked (p==0) keys: clamp in-bounds
        fbV.u[i] = *(const unsigned int*)&vrow[sc];
      }
      oacc[nt] = __builtin_amdgcn_wmma_f32_16x16x32_bf16(false, faP.v, false, fbV.v,
                                                         (short)0, oacc[nt], false, false);
    }
  }
  // ---- epilogue: normalize + head gate; bf16 out feeds next GEMM ----
  #pragma unroll
  for (int r = 0; r < 8; ++r) {
    int m = qb + r + half * 8;
    float g   = gates[((size_t)b * S_ + m) * H_ + h];
    float inv = 1.0f / rsum[r];
    #pragma unroll
    for (int nt = 0; nt < 4; ++nt) {
      int d = nt * 16 + l15;
      out[IDX(b, m, d)] = f2bf(oacc[nt][r] * inv * g);
    }
  }
#undef IDX
}

// ---------------- RMSNorm (one row per block), fp32 or bf16 out ----------------
template <int BF16OUT>
__global__ __launch_bounds__(256)
void rmsnorm_k(const float* __restrict__ x, const float* __restrict__ w,
               void* __restrict__ outp, int dim)
{
  __shared__ float red[8];
  const int row = blockIdx.x;
  const float* xr = x + (size_t)row * dim;
  float ss = 0.0f;
  for (int i = threadIdx.x; i < dim; i += 256) { float vv = xr[i]; ss += vv * vv; }
  #pragma unroll
  for (int sh = 16; sh >= 1; sh >>= 1) ss += __shfl_xor(ss, sh, 32);
  if ((threadIdx.x & 31) == 0) red[threadIdx.x >> 5] = ss;
  __syncthreads();
  float tot = 0.0f;
  #pragma unroll
  for (int i = 0; i < 8; ++i) tot += red[i];
  float rs = rsqrtf(tot / (float)dim + EPS_);
  for (int i = threadIdx.x; i < dim; i += 256) {
    float o = xr[i] * rs * w[i];
    if (BF16OUT) ((unsigned short*)outp)[(size_t)row * dim + i] = f2bf(o);
    else         ((float*)outp)[(size_t)row * dim + i] = o;
  }
}

// ---------------- RoPE (q,k) + value-residual mix; bf16 outputs ----------------
// qout/kout: [B,S,H,DH] bf16 ; vtout: [B,H,DH,S] bf16 (transposed) ; vres fp32
__global__ __launch_bounds__(256)
void rope_mix_k(const float* __restrict__ qin, const float* __restrict__ kv,
                const float* __restrict__ mix, unsigned short* __restrict__ qout,
                unsigned short* __restrict__ kout, unsigned short* __restrict__ vtout,
                float* __restrict__ vres, int layer)
{
  int t = blockIdx.x * 256 + threadIdx.x;
  int p = t & 31;   t >>= 5;   // DH/2 = 32 pairs
  int h = t & 15;   t >>= 4;
  int s = t & 2047; t >>= 11;
  int b = t;
  if (b >= B_) return;
  float ang = (float)s * __powf(10000.0f, -(float)(2 * p) / (float)DH_);
  float sn, c; __sincosf(ang, &sn, &c);
  size_t base = (((size_t)b * S_ + s) * H_ + h) * DH_;
  size_t kvb  = ((size_t)b * S_ + s) * (size_t)(2 * HD_) + (size_t)h * DH_;

  float q0 = qin[base + 2 * p], q1 = qin[base + 2 * p + 1];
  qout[base + 2 * p]     = f2bf((q0 * c - q1 * sn) * SCALE_);
  qout[base + 2 * p + 1] = f2bf((q1 * c + q0 * sn) * SCALE_);

  float k0 = kv[kvb + 2 * p], k1 = kv[kvb + 2 * p + 1];
  kout[base + 2 * p]     = f2bf(k0 * c - k1 * sn);
  kout[base + 2 * p + 1] = f2bf(k1 * c + k0 * sn);

  float v0 = kv[kvb + HD_ + 2 * p], v1 = kv[kvb + HD_ + 2 * p + 1];
  if (layer == 0) {
    vres[base + 2 * p] = v0;  vres[base + 2 * p + 1] = v1;
  } else {
    float mx = mix[((size_t)b * S_ + s) * H_ + h];
    v0 = v0 + mx * (vres[base + 2 * p]     - v0);
    v1 = v1 + mx * (vres[base + 2 * p + 1] - v1);
  }
  size_t vb = ((size_t)b * H_ + h) * DH_ * S_ + s;     // [b][h][d][s]
  vtout[vb + (size_t)(2 * p) * S_]     = f2bf(v0);
  vtout[vb + (size_t)(2 * p + 1) * S_] = f2bf(v1);
}

// ---------------- GLU: a * gelu_exact(g) ; bf16 out, K-padded to KP2_ ------
__global__ __launch_bounds__(256)
void glu_k(const float* __restrict__ f1, unsigned short* __restrict__ g2)
{
  int t = blockIdx.x * 256 + threadIdx.x;
  if (t >= M_ * KP2_) return;
  int row = t / KP2_;
  int c   = t - row * KP2_;
  float o = 0.0f;
  if (c < DIN_) {
    const float* fr = f1 + (size_t)row * (2 * DIN_);
    float a = fr[c];
    float g = fr[DIN_ + c];
    o = a * (0.5f * g * (1.0f + erff(g * 0.70710678118f)));
  }
  g2[(size_t)row * KP2_ + c] = f2bf(o);
}

// ---------------- orchestration ----------------
static inline int roundup(int v, int m) { return (v + m - 1) / m * m; }

extern "C" void kernel_launch(void* const* d_in, const int* in_sizes, int n_in,
                              void* d_out, int out_size, void* d_ws, size_t ws_size,
                              hipStream_t stream)
{
  const float* x_in  = (const float*)d_in[0];
  const float* anw   = (const float*)d_in[1];
  const float* wq    = (const float*)d_in[2];
  const float* wkv   = (const float*)d_in[3];
  const float* wo    = (const float*)d_in[4];
  const float* wg    = (const float*)d_in[5];
  const float* wmix  = (const float*)d_in[6];
  const float* ffnw  = (const float*)d_in[7];
  const float* ffinw = (const float*)d_in[8];
  const float* w1    = (const float*)d_in[9];
  const float* b1    = (const float*)d_in[10];
  const float* w2    = (const float*)d_in[11];
  const float* b2    = (const float*)d_in[12];
  const float* finw  = (const float*)d_in[13];
  float* out = (float*)d_out;

  char* base = (char*)d_ws;
  size_t off = 0;
  auto allocB = [&](size_t bytes) -> void* {
    off = (off + 255) & ~(size_t)255;
    void* p = base + off;
    off += bytes;
    return p;
  };
  const size_t MD = (size_t)M_ * DIM_;

  // fp32 buffers
  float* xw    = (float*)allocB(MD * 4);                    // running residual
  float* qbuf  = (float*)allocB(MD * 4);                    // q (pre-rope)
  float* kvbuf = (float*)allocB(MD * 2 * 4);                // [k|v]
  float* vres  = (float*)allocB(MD * 4);                    // layer-0 original v
  float* fbuf  = (float*)allocB(MD * 4);                    // ff first norm
  float* f1buf = (float*)allocB((size_t)M_ * 2 * DIN_ * 4); // w1 output
  float* gateb = (float*)allocB((size_t)M_ * H_ * 4);
  float* mixb  = (float*)allocB((size_t)M_ * H_ * 4);
  // bf16 buffers
  unsigned short* hb16 = (unsigned short*)allocB(MD * 2);   // rmsnorm(x)
  unsigned short* fb16 = (unsigned short*)allocB(MD * 2);   // ff double-norm
  unsigned short* ab16 = (unsigned short*)allocB(MD * 2);   // attention out
  unsigned short* qh   = (unsigned short*)allocB(MD * 2);   // roped q
  unsigned short* kh   = (unsigned short*)allocB(MD * 2);   // roped k
  unsigned short* vth  = (unsigned short*)allocB(MD * 2);   // v transposed
  unsigned short* g2b  = (unsigned short*)allocB((size_t)M_ * KP2_ * 2);
  // bf16 transposed weights (per-layer contiguous)
  const int NPQ = 1024, NPKV = 2048, NPO = 1024, NPG = 128, NPW1 = roundup(2 * DIN_, 128);
  unsigned short* wqT   = (unsigned short*)allocB((size_t)L_ * NPQ  * 1024 * 2);
  unsigned short* wkvT  = (unsigned short*)allocB((size_t)L_ * NPKV * 1024 * 2);
  unsigned short* woT   = (unsigned short*)allocB((size_t)L_ * NPO  * 1024 * 2);
  unsigned short* wgT   = (unsigned short*)allocB((size_t)L_ * NPG  * 1024 * 2);
  unsigned short* wmixT = (unsigned short*)allocB((size_t)L_ * NPG  * 1024 * 2);
  unsigned short* w1T   = (unsigned short*)allocB((size_t)L_ * NPW1 * 1024 * 2);
  unsigned short* w2T   = (unsigned short*)allocB((size_t)L_ * NPO  * KP2_ * 2);

  hipMemcpyAsync(xw, x_in, MD * 4, hipMemcpyDeviceToDevice, stream);

  dim3 blk(256);
  auto cgrid = [](int Kpad, int Npad) { return dim3(Kpad / 32, Npad / 32); };
  // one-time weight conversion (runs each launch; deterministic)
  for (int l = 0; l < L_; ++l) {
    convert_wT_k<<<cgrid(1024, NPQ),  blk, 0, stream>>>(wq   + (size_t)l*DIM_*HD_,     wqT   + (size_t)l*NPQ*1024,  DIM_, HD_,      1024);
    convert_wT_k<<<cgrid(1024, NPKV), blk, 0, stream>>>(wkv  + (size_t)l*DIM_*2*HD_,   wkvT  + (size_t)l*NPKV*1024, DIM_, 2*HD_,    1024);
    convert_wT_k<<<cgrid(1024, NPO),  blk, 0, stream>>>(wo   + (size_t)l*HD_*DIM_,     woT   + (size_t)l*NPO*1024,  HD_,  DIM_,     1024);
    convert_wT_k<<<cgrid(1024, NPG),  blk, 0, stream>>>(wg   + (size_t)l*DIM_*H_,      wgT   + (size_t)l*NPG*1024,  DIM_, H_,       1024);
    convert_wT_k<<<cgrid(1024, NPG),  blk, 0, stream>>>(wmix + (size_t)l*DIM_*H_,      wmixT + (size_t)l*NPG*1024,  DIM_, H_,       1024);
    convert_wT_k<<<cgrid(1024, NPW1), blk, 0, stream>>>(w1   + (size_t)l*DIM_*2*DIN_,  w1T   + (size_t)l*NPW1*1024, DIM_, 2*DIN_,   1024);
    convert_wT_k<<<cgrid(KP2_, NPO),  blk, 0, stream>>>(w2   + (size_t)l*DIN_*DIM_,    w2T   + (size_t)l*NPO*KP2_,  DIN_, DIM_,     KP2_);
  }

  auto gg = [](int N, int M) { return dim3((N + GBN - 1) / GBN, M / GBM); };

  for (int l = 0; l < L_; ++l) {
    const unsigned short* wqT_l   = wqT   + (size_t)l * NPQ  * 1024;
    const unsigned short* wkvT_l  = wkvT  + (size_t)l * NPKV * 1024;
    const unsigned short* woT_l   = woT   + (size_t)l * NPO  * 1024;
    const unsigned short* wgT_l   = wgT   + (size_t)l * NPG  * 1024;
    const unsigned short* wmixT_l = wmixT + (size_t)l * NPG  * 1024;
    const unsigned short* w1T_l   = w1T   + (size_t)l * NPW1 * 1024;
    const unsigned short* w2T_l   = w2T   + (size_t)l * NPO  * KP2_;
    const float* b1_l = b1 + (size_t)l * 2 * DIN_;
    const float* b2_l = b2 + (size_t)l * DIM_;

    rmsnorm_k<1><<<M_, blk, 0, stream>>>(xw, anw + (size_t)l * DIM_, hb16, DIM_);

    gemm_bf16<0><<<gg(HD_, M_), blk, 0, stream>>>(hb16, wqT_l,  nullptr, nullptr, qbuf,  M_, HD_,    DIM_, DIM_, 1024);
    gemm_bf16<0><<<gg(2*HD_, M_), blk, 0, stream>>>(hb16, wkvT_l, nullptr, nullptr, kvbuf, M_, 2*HD_, DIM_, DIM_, 1024);
    gemm_bf16<4><<<gg(H_, M_), blk, 0, stream>>>(hb16, wgT_l,  nullptr, nullptr, gateb, M_, H_,    DIM_, DIM_, 1024);
    if (l > 0)
      gemm_bf16<4><<<gg(H_, M_), blk, 0, stream>>>(hb16, wmixT_l, nullptr, nullptr, mixb, M_, H_,  DIM_, DIM_, 1024);

    {
      int total = B_ * S_ * H_ * (DH_ / 2);
      rope_mix_k<<<(total + 255) / 256, blk, 0, stream>>>(qbuf, kvbuf, mixb, qh, kh, vth, vres, l);
    }

    attn_wmma_k<<<dim3(S_ / 16, B_ * H_), dim3(32), 0, stream>>>(qh, kh, vth, gateb, ab16);

    gemm_bf16<2><<<gg(DIM_, M_), blk, 0, stream>>>(ab16, woT_l, nullptr, xw, xw, M_, DIM_, HD_, DIM_, 1024);

    rmsnorm_k<0><<<M_, blk, 0, stream>>>(xw, ffnw + (size_t)l * DIM_, fbuf, DIM_);
    rmsnorm_k<1><<<M_, blk, 0, stream>>>(fbuf, ffinw + (size_t)l * DIM_, fb16, DIM_);

    gemm_bf16<1><<<gg(2 * DIN_, M_), blk, 0, stream>>>(fb16, w1T_l, b1_l, nullptr, f1buf, M_, 2 * DIN_, DIM_, DIM_, 1024);

    {
      int total = M_ * KP2_;
      glu_k<<<(total + 255) / 256, blk, 0, stream>>>(f1buf, g2b);
    }

    gemm_bf16<3><<<gg(DIM_, M_), blk, 0, stream>>>(g2b, w2T_l, b2_l, xw, xw, M_, DIM_, DIN_, KP2_, KP2_);
  }

  rmsnorm_k<0><<<M_, blk, 0, stream>>>(xw, finw, out, DIM_);

  (void)in_sizes; (void)n_in; (void)out_size; (void)ws_size;
}
